// Attention_90967407330064
// MI455X (gfx1250) — compile-verified, matrix operand order checked by reading, not run on hardware
//
#include <hip/hip_runtime.h>

typedef __attribute__((ext_vector_type(16))) _Float16 v16h;
typedef __attribute__((ext_vector_type(8)))  _Float16 v8h;
typedef __attribute__((ext_vector_type(8)))  float    v8f;

#define B_DIM   16
#define C_DIM   256
#define N_DIM   4096
#define DA_DIM  64
#define XH_STRIDE 264   // 256 + 8 pad (f16): 16B-aligned rows, staggered banks
#define VT_STRIDE 72    // 64 + 8 pad
#define KT_STRIDE 72
#define PT_STRIDE 72

union F16Frag { v16h v; v8h h[2]; };

// Load a 16x32 f16 A/B WMMA fragment from token-major storage:
// lane&15 selects row, lane>>4 selects K-half; elements 0..7 = K {kb+h8..},
// elements 8..15 = K {kb+16+h8..} per ISA 7.12.2 layout. Two 16B loads.
__device__ __forceinline__ v16h load_frag(const _Float16* p0, int stride, int kb, int lane) {
    int h8 = (lane >> 4) << 3;       // 0 or 8
    int r  = lane & 15;
    const _Float16* p = p0 + (size_t)r * stride + kb;
    F16Frag u;
    u.h[0] = *(const v8h*)(p + h8);
    u.h[1] = *(const v8h*)(p + 16 + h8);
    return u.v;
}

// Same fragment but converting from f32 source (weight matrices).
__device__ __forceinline__ v16h load_frag_f32(const float* p0, int stride, int kb, int lane) {
    int h8 = (lane >> 4) << 3;
    int r  = lane & 15;
    const float* p = p0 + (size_t)r * stride + kb;
    F16Frag u;
#pragma unroll
    for (int i = 0; i < 8; ++i) u.v[i]     = (_Float16)p[h8 + i];
#pragma unroll
    for (int i = 0; i < 8; ++i) u.v[8 + i] = (_Float16)p[16 + h8 + i];
    return u.v;
}

// Async memory -> LDS copy, 16B per lane, tracked by ASYNCcnt (gfx1250).
__device__ __forceinline__ void async_ld16(unsigned lds_off, const _Float16* g) {
    asm volatile("global_load_async_to_lds_b128 %0, %1, off"
                 :: "v"(lds_off), "v"((unsigned long long)(uintptr_t)g)
                 : "memory");
}
__device__ __forceinline__ void wait_async0() {
    asm volatile("s_wait_asynccnt 0x0" ::: "memory");
}

// ---------------------------------------------------------------------------
// Kernel 1: QKV projection.
//   q = (Wq x + bq) * (1/64) -> qh [B][N][64] f16 (token-major; scale folded)
//   k =  Wk x + bk           -> kh [B][N][64] f16 (token-major)
//   v =  Wv x + bv           -> vh [B][C][N]  f16 (channel-major)
// One workgroup = (b, 64-token tile), 8 waves; 24 row-tiles of 16 output rows.
// ---------------------------------------------------------------------------
__global__ __launch_bounds__(256) void qkv_proj_kernel(
    const float* __restrict__ x,
    const float* __restrict__ Wq, const float* __restrict__ bq,
    const float* __restrict__ Wk, const float* __restrict__ bk,
    const float* __restrict__ Wv, const float* __restrict__ bv,
    _Float16* __restrict__ qh, _Float16* __restrict__ kh, _Float16* __restrict__ vh)
{
    __shared__ _Float16 xh[64 * XH_STRIDE];     // x tile, token-major [n][c], f16

    int b  = blockIdx.x >> 6;
    int n0 = (blockIdx.x & 63) * 64;
    int tid = threadIdx.x;

    const float* xb = x + (size_t)b * C_DIM * N_DIM;
#pragma unroll
    for (int e = 0; e < 64; ++e) {              // 256 threads x 64 = 16384 elems
        int idx = e * 256 + tid;
        int c = idx >> 6, n = idx & 63;
        xh[n * XH_STRIDE + c] = (_Float16)xb[(size_t)c * N_DIM + n0 + n];
    }
    __syncthreads();

    int wave = tid >> 5, lane = tid & 31;
    int col = lane & 15, hh = lane >> 4;

#pragma unroll
    for (int i = 0; i < 3; ++i) {
        int rt = wave * 3 + i;                  // 0..23
        const float* W;  const float* bias;  int rbase;  float sc;
        if (rt < 4)       { W = Wq; bias = bq; rbase = rt * 16;       sc = 0.015625f; }
        else if (rt < 8)  { W = Wk; bias = bk; rbase = (rt - 4) * 16; sc = 1.0f; }
        else              { W = Wv; bias = bv; rbase = (rt - 8) * 16; sc = 1.0f; }

        v8f acc[4] = {};
#pragma unroll
        for (int s = 0; s < 8; ++s) {           // K = 256 = 8 x 32
            v16h a = load_frag_f32(W + (size_t)rbase * C_DIM, C_DIM, s * 32, lane);
#pragma unroll
            for (int nt = 0; nt < 4; ++nt) {
                v16h bf = load_frag(xh + nt * 16 * XH_STRIDE, XH_STRIDE, s * 32, lane);
                acc[nt] = __builtin_amdgcn_wmma_f32_16x16x32_f16(
                    false, a, false, bf, (short)0, acc[nt], false, false);
            }
        }

        if (rt < 8) {                           // Q/K: [b][n][d], d contiguous -> 16B store
            _Float16* outp = (rt < 4) ? qh : kh;
            int dbase = rbase + 8 * hh;
#pragma unroll
            for (int nt = 0; nt < 4; ++nt) {
                int n = n0 + nt * 16 + col;
                v8h pk;
#pragma unroll
                for (int r = 0; r < 8; ++r)
                    pk[r] = (_Float16)((acc[nt][r] + bias[dbase + r]) * sc);
                *(v8h*)(outp + ((size_t)b * N_DIM + n) * DA_DIM + dbase) = pk;
            }
        } else {                                // V: [b][c][n], scatter per channel
            int cbase = rbase + 8 * hh;
#pragma unroll
            for (int nt = 0; nt < 4; ++nt) {
                int n = n0 + nt * 16 + col;
#pragma unroll
                for (int r = 0; r < 8; ++r)
                    vh[((size_t)b * C_DIM + cbase + r) * N_DIM + n] =
                        (_Float16)(acc[nt][r] + bias[cbase + r]);
            }
        }
    }
}

// ---------------------------------------------------------------------------
// Kernel 2: flash attention + residual, async double-buffered K/V tiles.
// Workgroup = (b, 64-query tile), 4 waves; wave owns 16 query rows x C=256 acc.
// Per key tile: S = Q K^T (WMMA), online softmax (shfl within 16-lane halves
// per C/D layout), P via LDS into A-layout, O += P V (WMMA). Tile j+1 is
// fetched with global_load_async_to_lds_b128 while tile j computes.
// ---------------------------------------------------------------------------
__global__ __launch_bounds__(128) void flash_attn_kernel(
    const float* __restrict__ x,
    const _Float16* __restrict__ qh, const _Float16* __restrict__ kh,
    const _Float16* __restrict__ vh, float* __restrict__ out)
{
    __shared__ _Float16 vt[2][C_DIM * VT_STRIDE];   // V tile [c][j] (transposed)
    __shared__ _Float16 kt[2][64 * KT_STRIDE];      // K tile [j][d] (token-major)
    __shared__ _Float16 pt[4 * 16 * PT_STRIDE];     // per-wave P tile [m][j]

    int b  = blockIdx.x >> 6;
    int i0 = (blockIdx.x & 63) * 64;
    int tid = threadIdx.x;
    int wave = tid >> 5, lane = tid & 31;
    int col = lane & 15, hh = lane >> 4;
    int m0 = i0 + wave * 16;

    const _Float16* vb = vh + (size_t)b * C_DIM * N_DIM;
    const _Float16* kb = kh + (size_t)b * N_DIM * DA_DIM;

    // persistent Q fragments (16 rows x 64 dims), scale already folded in
    const _Float16* qbase = qh + ((size_t)b * N_DIM + m0) * DA_DIM;
    v16h qf0 = load_frag(qbase, DA_DIM, 0,  lane);
    v16h qf1 = load_frag(qbase, DA_DIM, 32, lane);

    v8f acc[16] = {};
    float mst[8], lst[8];
#pragma unroll
    for (int r = 0; r < 8; ++r) { mst[r] = -1e30f; lst[r] = 0.f; }

    _Float16* ptw = pt + wave * 16 * PT_STRIDE;

    // prologue: issue async copies for tile 0
    {
        _Float16* vtb = vt[0]; _Float16* ktb = kt[0];
#pragma unroll
        for (int e = 0; e < 16; ++e) {          // V: 2048 x 16B segs / 128 thr
            int seg = e * 128 + tid;
            int c = seg >> 3, off = (seg & 7) * 8;
            async_ld16((unsigned)(uintptr_t)(vtb + c * VT_STRIDE + off),
                       vb + (size_t)c * N_DIM + off);
        }
#pragma unroll
        for (int e = 0; e < 4; ++e) {           // K: 512 x 16B segs / 128 thr
            int seg = e * 128 + tid;
            int j = seg >> 3, off = (seg & 7) * 8;
            async_ld16((unsigned)(uintptr_t)(ktb + j * KT_STRIDE + off),
                       kb + (size_t)j * DA_DIM + off);
        }
    }

    for (int jt = 0; jt < 64; ++jt) {
        int cur = jt & 1;
        wait_async0();                          // this wave's tile-jt copies done
        __syncthreads();                        // all waves done loading AND done
                                                // reading buffer (jt-1) == (jt+1)
        if (jt + 1 < 64) {                      // prefetch tile jt+1 (overlaps WMMA)
            int j1 = (jt + 1) * 64;
            _Float16* vtb = vt[cur ^ 1]; _Float16* ktb = kt[cur ^ 1];
#pragma unroll
            for (int e = 0; e < 16; ++e) {
                int seg = e * 128 + tid;
                int c = seg >> 3, off = (seg & 7) * 8;
                async_ld16((unsigned)(uintptr_t)(vtb + c * VT_STRIDE + off),
                           vb + (size_t)c * N_DIM + j1 + off);
            }
#pragma unroll
            for (int e = 0; e < 4; ++e) {
                int seg = e * 128 + tid;
                int j = seg >> 3, off = (seg & 7) * 8;
                async_ld16((unsigned)(uintptr_t)(ktb + j * KT_STRIDE + off),
                           kb + ((size_t)j1 + j) * DA_DIM + off);
            }
        }

        const _Float16* vtc = vt[cur];
        const _Float16* ktc = kt[cur];

        // S = Q K^T   (16 x 64, contraction = 64)
        v8f sacc[4];
#pragma unroll
        for (int t = 0; t < 4; ++t) {
            v16h kf0 = load_frag(ktc + t * 16 * KT_STRIDE, KT_STRIDE, 0,  lane);
            v16h kf1 = load_frag(ktc + t * 16 * KT_STRIDE, KT_STRIDE, 32, lane);
            v8f z = {};
            z = __builtin_amdgcn_wmma_f32_16x16x32_f16(false, qf0, false, kf0, (short)0, z, false, false);
            z = __builtin_amdgcn_wmma_f32_16x16x32_f16(false, qf1, false, kf1, (short)0, z, false, false);
            sacc[t] = z;
        }

        // online softmax: row = r + 8*hh, shared across the 16 lanes of a half
#pragma unroll
        for (int r = 0; r < 8; ++r) {
            float mx = -1e30f;
#pragma unroll
            for (int t = 0; t < 4; ++t) mx = fmaxf(mx, sacc[t][r]);
            mx = fmaxf(mx, __shfl_xor(mx, 1));
            mx = fmaxf(mx, __shfl_xor(mx, 2));
            mx = fmaxf(mx, __shfl_xor(mx, 4));
            mx = fmaxf(mx, __shfl_xor(mx, 8));
            float mnew  = fmaxf(mst[r], mx);
            float alpha = __expf(mst[r] - mnew);
            float rsum = 0.f;
#pragma unroll
            for (int t = 0; t < 4; ++t) {
                float p = __expf(sacc[t][r] - mnew);
                sacc[t][r] = p;
                rsum += p;
            }
            rsum += __shfl_xor(rsum, 1);
            rsum += __shfl_xor(rsum, 2);
            rsum += __shfl_xor(rsum, 4);
            rsum += __shfl_xor(rsum, 8);
            lst[r] = lst[r] * alpha + rsum;
            mst[r] = mnew;
#pragma unroll
            for (int cn = 0; cn < 16; ++cn) acc[cn][r] *= alpha;
        }

        // P (D-layout) -> per-wave LDS [m][j] so it can be re-read in A-layout
#pragma unroll
        for (int t = 0; t < 4; ++t)
#pragma unroll
            for (int r = 0; r < 8; ++r)
                ptw[(r + 8 * hh) * PT_STRIDE + t * 16 + col] = (_Float16)sacc[t][r];

        // O += P V   (contract 64 keys; B frags from transposed V tile)
        v16h pf0 = load_frag(ptw, PT_STRIDE, 0,  lane);
        v16h pf1 = load_frag(ptw, PT_STRIDE, 32, lane);
#pragma unroll
        for (int cn = 0; cn < 16; ++cn) {
            v16h vf0 = load_frag(vtc + cn * 16 * VT_STRIDE, VT_STRIDE, 0,  lane);
            v16h vf1 = load_frag(vtc + cn * 16 * VT_STRIDE, VT_STRIDE, 32, lane);
            acc[cn] = __builtin_amdgcn_wmma_f32_16x16x32_f16(false, pf0, false, vf0, (short)0, acc[cn], false, false);
            acc[cn] = __builtin_amdgcn_wmma_f32_16x16x32_f16(false, pf1, false, vf1, (short)0, acc[cn], false, false);
        }
    }

    // epilogue: O/l + residual x ; out[b][c][n], 8 consecutive n per lane
    const float* xb = x   + (size_t)b * C_DIM * N_DIM;
    float*       ob = out + (size_t)b * C_DIM * N_DIM;
#pragma unroll
    for (int cn = 0; cn < 16; ++cn) {
        int c = cn * 16 + col;
        const float* xr = xb + (size_t)c * N_DIM + m0 + 8 * hh;
        float*       orp = ob + (size_t)c * N_DIM + m0 + 8 * hh;
        v8f o;
#pragma unroll
        for (int r = 0; r < 8; ++r) o[r] = acc[cn][r] / lst[r] + xr[r];
        *(v8f*)orp = o;
    }
}

extern "C" void kernel_launch(void* const* d_in, const int* in_sizes, int n_in,
                              void* d_out, int out_size, void* d_ws, size_t ws_size,
                              hipStream_t stream) {
    const float* x  = (const float*)d_in[0];
    const float* Wq = (const float*)d_in[1];
    const float* bq = (const float*)d_in[2];
    const float* Wk = (const float*)d_in[3];
    const float* bk = (const float*)d_in[4];
    const float* Wv = (const float*)d_in[5];
    const float* bv = (const float*)d_in[6];
    float* out = (float*)d_out;

    // workspace: qh (8MB) | kh (8MB) | vh (32MB), all f16
    _Float16* qh = (_Float16*)d_ws;
    _Float16* kh = qh + (size_t)B_DIM * N_DIM * DA_DIM;
    _Float16* vh = kh + (size_t)B_DIM * N_DIM * DA_DIM;

    dim3 grid(B_DIM * (N_DIM / 64));
    qkv_proj_kernel<<<grid, 256, 0, stream>>>(x, Wq, bq, Wk, bk, Wv, bv, qh, kh, vh);
    flash_attn_kernel<<<grid, 128, 0, stream>>>(x, qh, kh, vh, out);
}
